// Compositional_dot_Transformer_25451976196692
// MI455X (gfx1250) — matrix operand order repaired and checked by reading, not compiled, not run on hardware
//
#include <hip/hip_runtime.h>
#include <hip/hip_bf16.h>

// ---------------------------------------------------------------------------
// Compositional dot Transformer, MI455X (gfx1250), wave32 + WMMA f16 pipeline.
// 1x4 register-blocked WMMA GEMMs (16x64 per wave) for A-fragment reuse.
// ---------------------------------------------------------------------------

typedef __attribute__((ext_vector_type(16))) _Float16 v16h;
typedef __attribute__((ext_vector_type(8)))  _Float16 v8h;
typedef __attribute__((ext_vector_type(8)))  float    v8f;
typedef __attribute__((ext_vector_type(4)))  float    v4f;

#define BATCH   4
#define NTOK    2048
#define DIMF    512
#define NSEARCH 8
#define HDIM    64
#define MTOK    (BATCH * NTOK)          // 8192 tokens total

// ---------------------------------------------------------------------------
// small packing kernels
// ---------------------------------------------------------------------------
__global__ void cvt_f32_to_f16(const float* __restrict__ in,
                               _Float16* __restrict__ out, int n) {
  int i = blockIdx.x * 256 + threadIdx.x;
  if (i < n) out[i] = (_Float16)in[i];
}

// WT[n][k] = (f16) W[k][n]   (so B-operand K runs contiguous per lane)
__global__ void transpose_to_f16(const float* __restrict__ W,
                                 _Float16* __restrict__ WT, int K, int N) {
  int i = blockIdx.x * 256 + threadIdx.x;
  if (i < K * N) {
    int k = i / N, n = i % N;
    WT[(size_t)n * K + k] = (_Float16)W[i];
  }
}

// ---------------------------------------------------------------------------
// projection GEMM:  C[M=8192, 512] = (A_f16 @ WT^T + bias) * scale
// each wave: 16x64 strip (4 accumulators), A-fragment reused across 4 WMMAs.
// optional transposed store (for V: [b][feature][token]).
// ---------------------------------------------------------------------------
__global__ void proj_gemm(const _Float16* __restrict__ A,
                          const _Float16* __restrict__ WT,
                          const float* __restrict__ bias,
                          _Float16* __restrict__ Crow,   // row-major out (or null)
                          _Float16* __restrict__ CvT,    // [b][512][2048] out (or null)
                          float scale) {
  const int lane = threadIdx.x & 31;
  const int wave = threadIdx.x >> 5;
  const int job  = blockIdx.x * 4 + wave;      // 4096 jobs
  const int mt = job >> 3;                     // 512 row tiles
  const int ng = job & 7;                      // 8 col groups of 64
  const int row0 = mt * 16, col0 = ng * 64;
  const int kg = lane >> 4;

  const _Float16* aptr = A  + (size_t)(row0 + (lane & 15)) * DIMF;
  const _Float16* bptr = WT + (size_t)(col0 + (lane & 15)) * DIMF;

  v8f acc[4] = {};
  for (int k0 = 0; k0 < DIMF; k0 += 32) {
    __builtin_prefetch(aptr + k0 + 128, 0, 1);
    union { v16h v; v8h h[2]; } ua;
    ua.h[0] = *(const v8h*)(aptr + k0 + kg * 8);
    ua.h[1] = *(const v8h*)(aptr + k0 + 16 + kg * 8);
#pragma unroll
    for (int i = 0; i < 4; ++i) {
      union { v16h v; v8h h[2]; } ub;
      const _Float16* bp = bptr + (size_t)i * 16 * DIMF + k0 + kg * 8;
      ub.h[0] = *(const v8h*)(bp);
      ub.h[1] = *(const v8h*)(bp + 16);
      acc[i] = __builtin_amdgcn_wmma_f32_16x16x32_f16(false, ua.v, false, ub.v,
                                                      (short)0, acc[i], false, false);
    }
  }

  const int rbase = row0 + (kg << 3);
#pragma unroll
  for (int i = 0; i < 4; ++i) {
    const int col = col0 + i * 16 + (lane & 15);
    const float bb = bias ? bias[col] : 0.f;
    for (int j = 0; j < 8; ++j) {
      const int row = rbase + j;
      const float v = (acc[i][j] + bb) * scale;
      if (Crow) Crow[(size_t)row * DIMF + col] = (_Float16)v;
      if (CvT) {
        const int b = row >> 11, n = row & (NTOK - 1);
        CvT[((size_t)b * DIMF + col) * NTOK + n] = (_Float16)v;
      }
    }
  }
}

// ---------------------------------------------------------------------------
// score GEMM: per (b,s)  S[n,m] = q[b,n,s,:] . k[b,m,s,:]   (scale folded in q)
// each wave: 16x64 strip of the 2048x2048 score tile; K = 64 (2 WMMA steps).
// raw f32 scores written straight into the d_out score region.
// ---------------------------------------------------------------------------
__global__ void score_gemm(const _Float16* __restrict__ qh,
                           const _Float16* __restrict__ kh,
                           float* __restrict__ score) {
  const int lane = threadIdx.x & 31;
  const int wave = threadIdx.x >> 5;
  const long job = (long)blockIdx.x * 4 + wave;    // 131072 jobs
  const int bs  = (int)(job >> 12);                // 4096 jobs per (b,s)
  const int rem = (int)(job & 4095);
  const int mt = rem >> 5;                         // 128 row tiles
  const int ng = rem & 31;                         // 32 col groups of 64
  const int b = bs >> 3, s = bs & 7;
  const int row0 = mt * 16, col0 = ng * 64;
  const int kg = lane >> 4;

  const _Float16* aptr = qh + (size_t)(b * NTOK + row0 + (lane & 15)) * DIMF + s * HDIM;
  const _Float16* bptr = kh + (size_t)(b * NTOK + col0 + (lane & 15)) * DIMF + s * HDIM;

  v8f acc[4] = {};
#pragma unroll
  for (int k0 = 0; k0 < HDIM; k0 += 32) {
    union { v16h v; v8h h[2]; } ua;
    ua.h[0] = *(const v8h*)(aptr + k0 + kg * 8);
    ua.h[1] = *(const v8h*)(aptr + k0 + 16 + kg * 8);
#pragma unroll
    for (int i = 0; i < 4; ++i) {
      union { v16h v; v8h h[2]; } ub;
      const _Float16* bp = bptr + (size_t)i * 16 * DIMF + k0 + kg * 8;
      ub.h[0] = *(const v8h*)(bp);
      ub.h[1] = *(const v8h*)(bp + 16);
      acc[i] = __builtin_amdgcn_wmma_f32_16x16x32_f16(false, ua.v, false, ub.v,
                                                      (short)0, acc[i], false, false);
    }
  }

  float* orow = score + (size_t)bs * NTOK * NTOK;
  const int rbase = row0 + (kg << 3);
#pragma unroll
  for (int i = 0; i < 4; ++i) {
    const int col = col0 + i * 16 + (lane & 15);
    for (int j = 0; j < 8; ++j)
      orow[(size_t)(rbase + j) * NTOK + col] = acc[i][j];
  }
}

// ---------------------------------------------------------------------------
// in-place masked softmax over each score row (diagonal -> -inf -> 0)
// one 256-thread block (8 wave32) per row of 2048.
// ---------------------------------------------------------------------------
__global__ void softmax_mask(float* __restrict__ score) {
  const int row = blockIdx.x;              // 65536 rows = 32 * 2048
  const int n   = row & (NTOK - 1);
  float* p = score + (size_t)row * NTOK;
  const int tid = threadIdx.x, lane = tid & 31, w = tid >> 5;

  float vals[8];
  float mx = -INFINITY;
  for (int i = 0; i < 8; ++i) {
    const int c = tid + i * 256;
    float v = p[c];
    if (c == n) v = -INFINITY;
    vals[i] = v;
    mx = fmaxf(mx, v);
  }
  for (int o = 16; o; o >>= 1) mx = fmaxf(mx, __shfl_xor(mx, o, 32));
  __shared__ float redm[8], reds[8];
  if (lane == 0) redm[w] = mx;
  __syncthreads();
  if (w == 0) {
    float m2 = (lane < 8) ? redm[lane] : -INFINITY;
    for (int o = 4; o; o >>= 1) m2 = fmaxf(m2, __shfl_xor(m2, o, 32));
    if (lane == 0) redm[0] = m2;
  }
  __syncthreads();
  mx = redm[0];

  float sum = 0.f;
  for (int i = 0; i < 8; ++i) { vals[i] = __expf(vals[i] - mx); sum += vals[i]; }
  for (int o = 16; o; o >>= 1) sum += __shfl_xor(sum, o, 32);
  if (lane == 0) reds[w] = sum;
  __syncthreads();
  if (w == 0) {
    float s2 = (lane < 8) ? reds[lane] : 0.f;
    for (int o = 4; o; o >>= 1) s2 += __shfl_xor(s2, o, 32);
    if (lane == 0) reds[0] = s2;
  }
  __syncthreads();
  const float inv = 1.0f / reds[0];
  for (int i = 0; i < 8; ++i) p[tid + i * 256] = vals[i] * inv;
}

// ---------------------------------------------------------------------------
// P @ V GEMM (dominant, ~137 GF): per (b,s)  out1[n, 512] = P[b,s] @ Vflat[b]
// Each wave: 16x64 strip. A = f32 probabilities (from d_out) converted to f16
// fragments on the fly and reused across 4 WMMAs; B = transposed V.
// ---------------------------------------------------------------------------
__global__ void pv_gemm(const float* __restrict__ P,
                        const _Float16* __restrict__ vT,
                        _Float16* __restrict__ out1) {
  const int lane = threadIdx.x & 31;
  const int wave = threadIdx.x >> 5;
  const long job = (long)blockIdx.x * 4 + wave;    // 32768 jobs
  const int bs  = (int)(job >> 10);                // 1024 jobs per (b,s)
  const int rem = (int)(job & 1023);
  const int mt = rem >> 3;                         // 128 row tiles
  const int ng = rem & 7;                          // 8 col groups of 64
  const int b = bs >> 3, s = bs & 7;
  const int row0 = mt * 16, col0 = ng * 64;
  const int kg = lane >> 4;

  const float*    aptr = P  + ((size_t)bs * NTOK + row0 + (lane & 15)) * NTOK;
  const _Float16* bptr = vT + ((size_t)b * DIMF + col0 + (lane & 15)) * NTOK;

  v8f acc[4] = {};
  for (int k0 = 0; k0 < NTOK; k0 += 32) {
    __builtin_prefetch(aptr + k0 + 256, 0, 1);
    const float* a0 = aptr + k0 + kg * 8;
    const float* a1 = aptr + k0 + 16 + kg * 8;
    const v4f f0 = *(const v4f*)a0, f1 = *(const v4f*)(a0 + 4);
    const v4f f2 = *(const v4f*)a1, f3 = *(const v4f*)(a1 + 4);
    v16h av;
#pragma unroll
    for (int i = 0; i < 4; ++i) {
      av[i]      = (_Float16)f0[i];
      av[4 + i]  = (_Float16)f1[i];
      av[8 + i]  = (_Float16)f2[i];
      av[12 + i] = (_Float16)f3[i];
    }
#pragma unroll
    for (int i = 0; i < 4; ++i) {
      union { v16h v; v8h h[2]; } ub;
      const _Float16* bp = bptr + (size_t)i * 16 * NTOK + k0 + kg * 8;
      ub.h[0] = *(const v8h*)(bp);
      ub.h[1] = *(const v8h*)(bp + 16);
      acc[i] = __builtin_amdgcn_wmma_f32_16x16x32_f16(false, av, false, ub.v,
                                                      (short)0, acc[i], false, false);
    }
  }

  const int rbase = row0 + (kg << 3);
#pragma unroll
  for (int i = 0; i < 4; ++i) {
    const int col = col0 + i * 16 + (lane & 15);
    for (int j = 0; j < 8; ++j) {
      const int row = rbase + j;
      out1[(((size_t)(b * NTOK + row)) * NSEARCH + s) * DIMF + col] = (_Float16)acc[i][j];
    }
  }
}

// ---------------------------------------------------------------------------
// Stage 2 gating, one wave32 per (b,n,s):
//   t      = Wvk @ q_v                       (64x64 matvec via lane shuffles)
//   logits = out1[r,:] . t + q_v . bvk       (8 wave reductions)
//   v_score = softmax_r(logits)  -> d_out    ; out2 = sum_r v_score[r]*out1[r,:]
// ---------------------------------------------------------------------------
__global__ void stage2_gate(const _Float16* __restrict__ out1,
                            const _Float16* __restrict__ qvh,
                            const _Float16* __restrict__ wvkT,  // [h][v]
                            const float* __restrict__ bvk,
                            float* __restrict__ vs_out,
                            _Float16* __restrict__ out2) {
  const int lane = threadIdx.x & 31;
  const int wave = threadIdx.x >> 5;
  const int gid  = blockIdx.x * 8 + wave;   // (b*2048+n)*8+s, 65536 total
  const int s  = gid & 7;
  const int bn = gid >> 3;

  const _Float16* qv = qvh + (size_t)bn * DIMF + s * HDIM;
  const float q_lo = (float)qv[lane];
  const float q_hi = (float)qv[lane + 32];

  float qb = q_lo * bvk[lane] + q_hi * bvk[lane + 32];
  for (int o = 16; o; o >>= 1) qb += __shfl_xor(qb, o, 32);

  float t_lo = 0.f, t_hi = 0.f;                  // t[lane], t[lane+32]
  for (int h = 0; h < 64; ++h) {
    const float qh_ = (h < 32) ? __shfl(q_lo, h, 32) : __shfl(q_hi, h - 32, 32);
    t_lo += (float)wvkT[h * 64 + lane]      * qh_;
    t_hi += (float)wvkT[h * 64 + lane + 32] * qh_;
  }

  const _Float16* orow = out1 + (size_t)gid * DIMF;
  float logits[8], mx = -INFINITY;
  for (int r = 0; r < 8; ++r) {
    float part = (float)orow[r * 64 + lane] * t_lo +
                 (float)orow[r * 64 + lane + 32] * t_hi;
    for (int o = 16; o; o >>= 1) part += __shfl_xor(part, o, 32);
    part += qb;
    logits[r] = part;
    mx = fmaxf(mx, part);
  }
  float w8[8], sum = 0.f;
  for (int r = 0; r < 8; ++r) { w8[r] = __expf(logits[r] - mx); sum += w8[r]; }
  const float inv = 1.0f / sum;
  if (lane < 8) vs_out[(size_t)gid * 8 + lane] = w8[lane] * inv;

  float o_lo = 0.f, o_hi = 0.f;
  for (int r = 0; r < 8; ++r) {
    const float wr = w8[r] * inv;
    o_lo += wr * (float)orow[r * 64 + lane];
    o_hi += wr * (float)orow[r * 64 + lane + 32];
  }
  out2[(size_t)bn * DIMF + s * HDIM + lane]      = (_Float16)o_lo;
  out2[(size_t)bn * DIMF + s * HDIM + lane + 32] = (_Float16)o_hi;
}

// ---------------------------------------------------------------------------
// final GEMM:  y[8192, 512] = out2 @ Wo + bo   (f32 output, 16x64 per wave)
// ---------------------------------------------------------------------------
__global__ void out_gemm(const _Float16* __restrict__ A,
                         const _Float16* __restrict__ WT,
                         const float* __restrict__ bias,
                         float* __restrict__ Y) {
  const int lane = threadIdx.x & 31;
  const int wave = threadIdx.x >> 5;
  const int job  = blockIdx.x * 4 + wave;
  const int mt = job >> 3, ng = job & 7;
  const int row0 = mt * 16, col0 = ng * 64;
  const int kg = lane >> 4;

  const _Float16* aptr = A  + (size_t)(row0 + (lane & 15)) * DIMF;
  const _Float16* bptr = WT + (size_t)(col0 + (lane & 15)) * DIMF;

  v8f acc[4] = {};
  for (int k0 = 0; k0 < DIMF; k0 += 32) {
    union { v16h v; v8h h[2]; } ua;
    ua.h[0] = *(const v8h*)(aptr + k0 + kg * 8);
    ua.h[1] = *(const v8h*)(aptr + k0 + 16 + kg * 8);
#pragma unroll
    for (int i = 0; i < 4; ++i) {
      union { v16h v; v8h h[2]; } ub;
      const _Float16* bp = bptr + (size_t)i * 16 * DIMF + k0 + kg * 8;
      ub.h[0] = *(const v8h*)(bp);
      ub.h[1] = *(const v8h*)(bp + 16);
      acc[i] = __builtin_amdgcn_wmma_f32_16x16x32_f16(false, ua.v, false, ub.v,
                                                      (short)0, acc[i], false, false);
    }
  }

  const int rbase = row0 + (kg << 3);
#pragma unroll
  for (int i = 0; i < 4; ++i) {
    const int col = col0 + i * 16 + (lane & 15);
    const float bb = bias[col];
    for (int j = 0; j < 8; ++j)
      Y[(size_t)(rbase + j) * DIMF + col] = acc[i][j] + bb;
  }
}

// ---------------------------------------------------------------------------
// host launcher
// ---------------------------------------------------------------------------
extern "C" void kernel_launch(void* const* d_in, const int* in_sizes, int n_in,
                              void* d_out, int out_size, void* d_ws, size_t ws_size,
                              hipStream_t stream) {
  (void)in_sizes; (void)n_in; (void)out_size; (void)ws_size;
  const float* x   = (const float*)d_in[0];
  const float* Wq  = (const float*)d_in[1];  const float* bq  = (const float*)d_in[2];
  const float* Wk  = (const float*)d_in[3];  const float* bk  = (const float*)d_in[4];
  const float* Wv  = (const float*)d_in[5];  const float* bv  = (const float*)d_in[6];
  const float* Wvq = (const float*)d_in[7];  const float* bvq = (const float*)d_in[8];
  const float* Wvk = (const float*)d_in[9];  const float* bvk = (const float*)d_in[10];
  const float* Wo  = (const float*)d_in[11]; const float* bo  = (const float*)d_in[12];

  // d_out: y [4,2048,512] | score [4,8,1,2048,2048] | v_score [4,2048,8,8,1]
  float* y_out  = (float*)d_out;
  float* sc_out = y_out + (size_t)MTOK * DIMF;
  float* vs_out = sc_out + (size_t)BATCH * NSEARCH * NTOK * NTOK;

  // workspace carve-up (all f16, all 4096-half aligned)
  _Float16* w16 = (_Float16*)d_ws;
  size_t off = 0;
  _Float16* xh    = w16 + off; off += (size_t)MTOK * DIMF;          // 4.2M
  _Float16* qh    = w16 + off; off += (size_t)MTOK * DIMF;
  _Float16* kh    = w16 + off; off += (size_t)MTOK * DIMF;
  _Float16* qvh   = w16 + off; off += (size_t)MTOK * DIMF;
  _Float16* out2h = w16 + off; off += (size_t)MTOK * DIMF;
  _Float16* vTh   = w16 + off; off += (size_t)BATCH * DIMF * NTOK;  // 4.2M
  _Float16* out1h = w16 + off; off += (size_t)MTOK * NSEARCH * DIMF;// 33.6M
  _Float16* wqT   = w16 + off; off += (size_t)DIMF * DIMF;
  _Float16* wkT   = w16 + off; off += (size_t)DIMF * DIMF;
  _Float16* wvT   = w16 + off; off += (size_t)DIMF * DIMF;
  _Float16* wvqT  = w16 + off; off += (size_t)DIMF * DIMF;
  _Float16* woT   = w16 + off; off += (size_t)DIMF * DIMF;
  _Float16* wvkT  = w16 + off; off += (size_t)HDIM * HDIM;

  const float scl = 0.125f;  // HD^-0.5

  // pack
  cvt_f32_to_f16<<<(MTOK * DIMF + 255) / 256, 256, 0, stream>>>(x, xh, MTOK * DIMF);
  transpose_to_f16<<<(DIMF * DIMF + 255) / 256, 256, 0, stream>>>(Wq,  wqT,  DIMF, DIMF);
  transpose_to_f16<<<(DIMF * DIMF + 255) / 256, 256, 0, stream>>>(Wk,  wkT,  DIMF, DIMF);
  transpose_to_f16<<<(DIMF * DIMF + 255) / 256, 256, 0, stream>>>(Wv,  wvT,  DIMF, DIMF);
  transpose_to_f16<<<(DIMF * DIMF + 255) / 256, 256, 0, stream>>>(Wvq, wvqT, DIMF, DIMF);
  transpose_to_f16<<<(DIMF * DIMF + 255) / 256, 256, 0, stream>>>(Wo,  woT,  DIMF, DIMF);
  transpose_to_f16<<<(HDIM * HDIM + 255) / 256, 256, 0, stream>>>(Wvk, wvkT, HDIM, HDIM);

  // projections (q/q_v carry the 1/sqrt(HD) scale)
  proj_gemm<<<1024, 128, 0, stream>>>(xh, wqT,  bq,  qh,  nullptr, scl);
  proj_gemm<<<1024, 128, 0, stream>>>(xh, wkT,  bk,  kh,  nullptr, 1.0f);
  proj_gemm<<<1024, 128, 0, stream>>>(xh, wvT,  bv,  nullptr, vTh, 1.0f);
  proj_gemm<<<1024, 128, 0, stream>>>(xh, wvqT, bvq, qvh, nullptr, scl);

  // stage 1 attention
  score_gemm<<<32768, 128, 0, stream>>>(qh, kh, sc_out);
  softmax_mask<<<65536, 256, 0, stream>>>(sc_out);
  pv_gemm<<<8192, 128, 0, stream>>>(sc_out, vTh, out1h);

  // stage 2 gating + output projection
  stage2_gate<<<8192, 256, 0, stream>>>(out1h, qvh, wvkT, bvk, vs_out, out2h);
  out_gemm<<<1024, 128, 0, stream>>>(out2h, woT, bo, y_out);
}
